// Prompt_23931557773363
// MI455X (gfx1250) — matrix operand order repaired and checked by reading, not compiled, not run on hardware
//
#include <hip/hip_runtime.h>
#include <math.h>

typedef __attribute__((ext_vector_type(16))) _Float16 v16h;
typedef __attribute__((ext_vector_type(8)))  float    v8f;

#define D_MODEL 6670
#define HID     100
#define B_SZ    32
#define S_SZ    256
#define NEXP    7
#define ROWS    (B_SZ * S_SZ)   // 8192
#define KC_DOWN 209             // ceil(6670/32)
#define NT_DOWN 7               // ceil(100/16)  (h padded 100->112)
#define DT_UP   417             // ceil(6670/16)
#define KC_UP   4               // K=100 padded to 128
#define YSTRIDE 128
#define DTB     8               // d-tiles per block in up-GEMM (128 cols)

// ---------------------------------------------------------------- WMMA helpers
__device__ __forceinline__ v8f wmma_f16(v16h a, v16h b, v8f c) {
  // v_wmma_f32_16x16x32_f16  (probe-confirmed 8-arg signature)
  return __builtin_amdgcn_wmma_f32_16x16x32_f16(false, a, false, b, (short)0, c,
                                                false, false);
}

// A-matrix 16x32 f16 layout: lane L holds row M=L%16,
// K = 8*(L>>4) + 16*(e>>3) + (e&7)  -> two contiguous runs of 8.
__device__ __forceinline__ v16h load_a_full(const float* p /* rowbase + kbl */) {
  v16h a;
  const float2* q0 = (const float2*)p;
  const float2* q1 = (const float2*)(p + 16);
#pragma unroll
  for (int i = 0; i < 4; ++i) {
    float2 u = q0[i];
    float2 w = q1[i];
    a[2 * i + 0] = (_Float16)u.x;
    a[2 * i + 1] = (_Float16)u.y;
    a[8 + 2 * i + 0] = (_Float16)w.x;
    a[8 + 2 * i + 1] = (_Float16)w.y;
  }
  return a;
}

__device__ __forceinline__ v16h load_a_guard(const float* rowbase, int kbl, int kmax) {
  v16h a;
#pragma unroll
  for (int e = 0; e < 16; ++e) {
    int k = kbl + ((e >> 3) << 4) + (e & 7);
    a[e] = (_Float16)((k < kmax) ? rowbase[k] : 0.0f);
  }
  return a;
}

// order-preserving float<->uint (for max via atomicMax on unsigned)
__device__ __forceinline__ unsigned fenc(float f) {
  unsigned u = __float_as_uint(f);
  return (u & 0x80000000u) ? ~u : (u | 0x80000000u);
}
__device__ __forceinline__ float fdec(unsigned u) {
  return (u & 0x80000000u) ? __uint_as_float(u & 0x7FFFFFFFu)
                           : __uint_as_float(~u);
}

__device__ __forceinline__ float silu(float z) { return z / (1.0f + __expf(-z)); }

// ------------------------------------------------- 1. f16 B-fragment prepacking
// B-matrix 32x16 f16 layout: lane L holds col N=L%16, K = 8*(L>>4) + 16*(e>>3) + (e&7).
__global__ void k_prep(const float* __restrict__ Wd, const float* __restrict__ Wu,
                       v16h* __restrict__ bdown, v16h* __restrict__ bup) {
  const int TD = NT_DOWN * KC_DOWN * 32;  // 46816
  const int TU = DT_UP * KC_UP * 32;      // 53376
  int id = blockIdx.x * 256 + threadIdx.x;
  if (id < TD) {
    int lane = id & 31, rest = id >> 5;
    int kc = rest % KC_DOWN, nt = rest / KC_DOWN;
    int h  = nt * 16 + (lane & 15);
    int kb = kc * 32 + ((lane >> 4) << 3);
    v16h v;
#pragma unroll
    for (int e = 0; e < 16; ++e) {
      int d = kb + ((e >> 3) << 4) + (e & 7);  // B[k=d][n=h] = Wd[h, d]
      v[e] = (_Float16)((h < HID && d < D_MODEL) ? Wd[(size_t)h * D_MODEL + d] : 0.0f);
    }
    bdown[id] = v;
  } else if (id < TD + TU) {
    int id2 = id - TD;
    int lane = id2 & 31, rest = id2 >> 5;
    int kc = rest & 3, dt = rest >> 2;
    int d  = dt * 16 + (lane & 15);
    int kb = kc * 32 + ((lane >> 4) << 3);
    v16h v;
#pragma unroll
    for (int e = 0; e < 16; ++e) {
      int h = kb + ((e >> 3) << 4) + (e & 7);  // B[k=h][n=d] = Wu[d, h]
      v[e] = (_Float16)((d < D_MODEL && h < HID) ? Wu[(size_t)d * HID + h] : 0.0f);
    }
    bup[id2] = v;
  }
}

__global__ void k_zero(float* __restrict__ p, int n) {
  int i = blockIdx.x * 256 + threadIdx.x;
  if (i < n) p[i] = 0.0f;
}

// -------------------------------------- 2. Gram = W_up^T W_up  and colsum(W_up)
__global__ void k_gram(const float* __restrict__ Wup, float* __restrict__ Gram,
                       float* __restrict__ colsum) {
  __shared__ float row[HID];
  int t = threadIdx.x;
  int d0 = blockIdx.x * 209;
  int d1 = (d0 + 209 < D_MODEL) ? d0 + 209 : D_MODEL;
  int active = (t < 200);
  int h1 = t % 100;
  int h2b = (t < 100) ? 0 : 50;
  float acc[50];
#pragma unroll
  for (int i = 0; i < 50; ++i) acc[i] = 0.0f;
  float cs = 0.0f;
  for (int d = d0; d < d1; ++d) {
    __syncthreads();
    if (t < HID) row[t] = Wup[(size_t)d * HID + t];
    __syncthreads();
    if (t < HID) cs += row[t];
    if (active) {
      float yh1 = row[h1];
#pragma unroll
      for (int i = 0; i < 50; ++i) acc[i] += yh1 * row[h2b + i];
    }
  }
  if (active)
    for (int i = 0; i < 50; ++i) atomicAdd(&Gram[h1 * HID + h2b + i], acc[i]);
  if (t < HID) atomicAdd(&colsum[t], cs);
}

// ------------------------------- 3. Y = silu(X @ W_down^T)  [8192 x 128 padded]
__global__ void k_gemm_down(const float* __restrict__ X, const v16h* __restrict__ bdown,
                            float* __restrict__ Y) {
  int lane = threadIdx.x & 31;
  int wave = blockIdx.x * (blockDim.x >> 5) + (threadIdx.x >> 5);
  int mbase = wave * 16;
  int row = mbase + (lane & 15);
  const float* rowbase = X + (size_t)row * D_MODEL;

  v8f acc[NT_DOWN];
#pragma unroll
  for (int nt = 0; nt < NT_DOWN; ++nt)
#pragma unroll
    for (int e = 0; e < 8; ++e) acc[nt][e] = 0.0f;

  for (int kc = 0; kc < KC_DOWN; ++kc) {
    int kbl = kc * 32 + ((lane >> 4) << 3);
    v16h a = (kc < KC_DOWN - 1) ? load_a_full(rowbase + kbl)
                                : load_a_guard(rowbase, kbl, D_MODEL);
#pragma unroll
    for (int nt = 0; nt < NT_DOWN; ++nt) {
      v16h b = bdown[(nt * KC_DOWN + kc) * 32 + lane];
      acc[nt] = wmma_f16(a, b, acc[nt]);
    }
  }
  // C/D layout: lane L: col = L%16, rows M = v + 8*(L>>4)
  int rbase = mbase + ((lane >> 4) << 3);
#pragma unroll
  for (int nt = 0; nt < NT_DOWN; ++nt) {
    int h = nt * 16 + (lane & 15);
#pragma unroll
    for (int v = 0; v < 8; ++v)
      Y[(size_t)(rbase + v) * YSTRIDE + h] = silu(acc[nt][v]);
  }
#pragma unroll
  for (int v = 0; v < 8; ++v)  // zero-pad cols 112..127
    Y[(size_t)(rbase + v) * YSTRIDE + 112 + (lane & 15)] = 0.0f;
}

// ------------------- 4. per-row mean/rstd of z via colsum & Gram quadratic form
__global__ void k_rowstats(const float* __restrict__ Y, const float* __restrict__ Gram,
                           const float* __restrict__ colsum, float2* __restrict__ mr) {
  __shared__ float sG[HID * 128];  // padded cols -> 51.2 KB
  __shared__ float yb[8][128];
  __shared__ float scs[128];
  int tid = threadIdx.x, lane = tid & 31, w = tid >> 5;
  for (int i = tid; i < HID * 128; i += 256) {
    int h2 = i & 127;
    sG[i] = (h2 < HID) ? Gram[(i >> 7) * HID + h2] : 0.0f;
  }
  if (tid < 128) scs[tid] = (tid < HID) ? colsum[tid] : 0.0f;
  int rowi = blockIdx.x * 8 + w;
  const float* yrow = Y + (size_t)rowi * YSTRIDE;
#pragma unroll
  for (int i = 0; i < 4; ++i) {
    int h = lane + 32 * i;
    yb[w][h] = (h < HID) ? yrow[h] : 0.0f;
  }
  __syncthreads();
  float t0 = yb[w][lane], t1 = yb[w][lane + 32];
  float t2 = yb[w][lane + 64], t3 = yb[w][lane + 96];
  float macc = t0 * scs[lane] + t1 * scs[lane + 32] + t2 * scs[lane + 64] +
               t3 * scs[lane + 96];
  float q = 0.0f;
  for (int h1 = 0; h1 < HID; ++h1) {
    float yh1 = yb[w][h1];
    const float* gr = &sG[h1 * 128];
    q += yh1 * (t0 * gr[lane] + t1 * gr[lane + 32] + t2 * gr[lane + 64] +
                t3 * gr[lane + 96]);
  }
  for (int o = 16; o > 0; o >>= 1) {
    q    += __shfl_xor(q, o, 32);
    macc += __shfl_xor(macc, o, 32);
  }
  if (lane == 0) {
    float mean = macc / (float)D_MODEL;
    float var = q / (float)D_MODEL - mean * mean;
    mr[rowi] = make_float2(mean, rsqrtf(var + 1e-5f));
  }
}

// -------- 5. up-GEMM fused with LayerNorm + max over seq -> xmax[32,D]
// Block = (8 d-tiles = 128 cols, one batch). B frags staged in LDS (32 KB),
// A frags held in registers and reused across all 8 d-tiles.
__global__ void k_up_ln_max(const float* __restrict__ Y, const v16h* __restrict__ bup,
                            const float2* __restrict__ mr, const float* __restrict__ gamma,
                            const float* __restrict__ beta, float* __restrict__ xmax) {
  __shared__ v16h sB[DTB * KC_UP * 32];   // 32 KB
  __shared__ unsigned smax[DTB * 16];
  int tid = threadIdx.x, lane = tid & 31, w = tid >> 5;
  int dt0 = blockIdx.x * DTB, b = blockIdx.y;

  // stage B fragments for the 8 d-tiles into LDS
  for (int i = tid; i < DTB * KC_UP * 32; i += 256) {
    int l = i & 31, kc = (i >> 5) & 3, dtl = i >> 7;
    int dt = dt0 + dtl;
    v16h v;
    if (dt < DT_UP) {
      v = bup[((size_t)dt * KC_UP + kc) * 32 + l];
    } else {
#pragma unroll
      for (int e = 0; e < 16; ++e) v[e] = (_Float16)0.0f;
    }
    sB[i] = v;
  }
  if (tid < DTB * 16) smax[tid] = 0u;

  int col = lane & 15, hh = lane >> 4;
  float g[DTB], be[DTB], wmax[DTB];
#pragma unroll
  for (int dtl = 0; dtl < DTB; ++dtl) {
    int d = (dt0 + dtl) * 16 + col;
    g[dtl]  = (d < D_MODEL) ? gamma[d] : 0.0f;
    be[dtl] = (d < D_MODEL) ? beta[d] : 0.0f;
    wmax[dtl] = -3.0e38f;
  }
  __syncthreads();

#pragma unroll
  for (int mt = 0; mt < 2; ++mt) {
    int sbase = w * 32 + mt * 16;
    const float* rowbase = Y + (size_t)(b * S_SZ + sbase + col) * YSTRIDE;
    v16h a4[KC_UP];
#pragma unroll
    for (int kc = 0; kc < KC_UP; ++kc)
      a4[kc] = load_a_full(rowbase + kc * 32 + (hh << 3));

    float2 m8[8];
#pragma unroll
    for (int v = 0; v < 8; ++v)
      m8[v] = mr[b * S_SZ + sbase + v + (hh << 3)];

#pragma unroll
    for (int dtl = 0; dtl < DTB; ++dtl) {
      v8f acc;
#pragma unroll
      for (int e = 0; e < 8; ++e) acc[e] = 0.0f;
#pragma unroll
      for (int kc = 0; kc < KC_UP; ++kc)
        acc = wmma_f16(a4[kc], sB[(dtl * KC_UP + kc) * 32 + lane], acc);
#pragma unroll
      for (int v = 0; v < 8; ++v)
        wmax[dtl] = fmaxf(wmax[dtl], (acc[v] - m8[v].x) * m8[v].y * g[dtl] + be[dtl]);
    }
  }
#pragma unroll
  for (int dtl = 0; dtl < DTB; ++dtl)
    atomicMax(&smax[dtl * 16 + col], fenc(wmax[dtl]));
  __syncthreads();
  if (tid < DTB * 16) {
    int dd = (dt0 + (tid >> 4)) * 16 + (tid & 15);
    if (dd < D_MODEL) xmax[(size_t)b * D_MODEL + dd] = fdec(smax[tid]);
  }
}

// ------------------------------------------- 6. gate scores + softmax -> sm[3,32,8]
__global__ void k_gate(const float* __restrict__ xmax,
                       const float* __restrict__ GA, const float* __restrict__ WA,
                       const float* __restrict__ GC, const float* __restrict__ WC,
                       const float* __restrict__ GE, const float* __restrict__ WE,
                       float* __restrict__ sm) {
  int k = blockIdx.x, b = blockIdx.y, tid = threadIdx.x;
  const float* G = (k == 0) ? GA : (k == 1) ? GC : GE;
  const float* W = (k == 0) ? WA : (k == 1) ? WC : WE;
  float acc[8];
#pragma unroll
  for (int j = 0; j < 8; ++j) acc[j] = 0.0f;
  const float* xb = xmax + (size_t)b * D_MODEL;
  for (int d = tid; d < D_MODEL; d += 256) {
    float xv = xb[d];
#pragma unroll
    for (int j = 0; j < NEXP; ++j) {
      const float* gp = G + ((size_t)j * D_MODEL + d) * 3;
      acc[j] += xv * (gp[0] + gp[1] + gp[2]);
    }
    const float* wp = W + (size_t)d * 3;
    acc[7] += xv * (wp[0] + wp[1] + wp[2]);
  }
  __shared__ float part[8][8];
  int lane = tid & 31, w = tid >> 5;
#pragma unroll
  for (int j = 0; j < 8; ++j) {
    float v = acc[j];
    for (int o = 16; o > 0; o >>= 1) v += __shfl_xor(v, o, 32);
    if (lane == 0) part[w][j] = v;
  }
  __syncthreads();
  if (tid == 0) {
    float s[8];
    for (int j = 0; j < 8; ++j) {
      float t = 0.0f;
      for (int ww = 0; ww < 8; ++ww) t += part[ww][j];
      s[j] = t * 0.125f;  // / TEMP
    }
    float mx = s[0];
    for (int j = 1; j < 8; ++j) mx = fmaxf(mx, s[j]);
    float es[8], tot = 0.0f;
    for (int j = 0; j < 8; ++j) { es[j] = __expf(s[j] - mx); tot += es[j]; }
    for (int j = 0; j < 8; ++j) sm[(k * B_SZ + b) * 8 + j] = es[j] / tot;
  }
}

// -------------------- 7. outs[i][b,d,c] = sum_j sm*G + W  -> d_out (concatenated)
__global__ void k_out(const float* __restrict__ GA, const float* __restrict__ WA,
                      const float* __restrict__ GC, const float* __restrict__ WC,
                      const float* __restrict__ GE, const float* __restrict__ WE,
                      const float* __restrict__ sm, float* __restrict__ out) {
  int i = blockIdx.z, b = blockIdx.y;
  int e = blockIdx.x * 256 + threadIdx.x;  // over D*3 = 20010
  const float* G = (i == 0) ? GA : (i == 1) ? GC : GE;
  const float* W = (i == 0) ? WA : (i == 1) ? WC : WE;
  __shared__ float wj[NEXP];
  if (threadIdx.x < NEXP) wj[threadIdx.x] = sm[(i * B_SZ + b) * 8 + threadIdx.x];
  __syncthreads();
  if (e >= D_MODEL * 3) return;
  float r = W[e];
#pragma unroll
  for (int j = 0; j < NEXP; ++j) r += wj[j] * G[(size_t)j * (D_MODEL * 3) + e];
  out[((size_t)i * B_SZ + b) * (D_MODEL * 3) + e] = r;
}

// ------------------------------------------------------------------- launcher
extern "C" void kernel_launch(void* const* d_in, const int* in_sizes, int n_in,
                              void* d_out, int out_size, void* d_ws, size_t ws_size,
                              hipStream_t stream) {
  const float* X      = (const float*)d_in[0];
  const float* W_A    = (const float*)d_in[1];
  const float* G_A    = (const float*)d_in[2];
  const float* W_C    = (const float*)d_in[3];
  const float* G_C    = (const float*)d_in[4];
  const float* W_E    = (const float*)d_in[5];
  const float* G_E    = (const float*)d_in[6];
  const float* W_down = (const float*)d_in[7];
  const float* W_up   = (const float*)d_in[8];
  const float* gamma  = (const float*)d_in[9];
  const float* beta   = (const float*)d_in[10];

  char* ws = (char*)d_ws;
  size_t off = 0;
  v16h* bdown = (v16h*)(ws + off); off += (size_t)NT_DOWN * KC_DOWN * 32 * 32; // 1,498,112
  v16h* bup   = (v16h*)(ws + off); off += (size_t)DT_UP * KC_UP * 32 * 32;     // 1,708,032
  float* Y    = (float*)(ws + off); off += (size_t)ROWS * YSTRIDE * 4;         // 4,194,304
  float* Gram = (float*)(ws + off); off += (size_t)HID * HID * 4;              // 40,000
  float* csum = (float*)(ws + off); off += 512;                                // 128 floats
  float2* mr  = (float2*)(ws + off); off += (size_t)ROWS * 8;
  float* xmax = (float*)(ws + off); off += (size_t)B_SZ * D_MODEL * 4;
  float* sm   = (float*)(ws + off); off += (size_t)3 * B_SZ * 8 * 4;
  (void)off; (void)ws_size; (void)in_sizes; (void)n_in; (void)out_size;

  const int TTOT = NT_DOWN * KC_DOWN * 32 + DT_UP * KC_UP * 32;  // 100192
  k_prep<<<(TTOT + 255) / 256, 256, 0, stream>>>(W_down, W_up, bdown, bup);
  k_zero<<<(HID * HID + 128 + 255) / 256, 256, 0, stream>>>(Gram, HID * HID + 128);
  k_gram<<<32, 256, 0, stream>>>(W_up, Gram, csum);
  k_gemm_down<<<(ROWS / 16) / 8, 256, 0, stream>>>(X, bdown, Y);           // 64 blocks
  k_rowstats<<<ROWS / 8, 256, 0, stream>>>(Y, Gram, csum, mr);             // 1024 blocks
  k_up_ln_max<<<dim3((DT_UP + DTB - 1) / DTB, B_SZ), 256, 0, stream>>>(
      Y, bup, mr, gamma, beta, xmax);                                       // (53, 32)
  k_gate<<<dim3(3, B_SZ), 256, 0, stream>>>(xmax, G_A, W_A, G_C, W_C, G_E, W_E, sm);
  k_out<<<dim3((D_MODEL * 3 + 255) / 256, B_SZ, 3), 256, 0, stream>>>(
      G_A, W_A, G_C, W_C, G_E, W_E, sm, (float*)d_out);
}
